// GeoBiasEncoderLayer_31396210934321
// MI455X (gfx1250) — matrix operand order repaired and checked by reading, not compiled, not run on hardware
//
#include <hip/hip_runtime.h>
#include <hip/hip_bf16.h>

typedef __attribute__((ext_vector_type(16))) __bf16 v16bf;
typedef __attribute__((ext_vector_type(8)))  float  v8f;
typedef __attribute__((ext_vector_type(4)))  unsigned int v4u;
typedef __attribute__((ext_vector_type(8)))  int v8i_;
typedef __attribute__((ext_vector_type(4)))  int v4i_;
typedef unsigned short ushort_t;

#define D_MODEL 768
#define N_HEADS 12
#define D_K     64
#define D_FF    3072

#if defined(__has_builtin)
#if __has_builtin(__builtin_amdgcn_tensor_load_to_lds) && __has_builtin(__builtin_amdgcn_s_wait_tensorcnt)
#define HAVE_TDM 1
#endif
#endif
#ifndef HAVE_TDM
#define HAVE_TDM 0
#endif

// ---------------- helpers ----------------

__device__ __forceinline__ unsigned short f2bf_u(float f) {
    unsigned int u = __float_as_uint(f);
    u += 0x7fffu + ((u >> 16) & 1u);   // round to nearest even
    return (unsigned short)(u >> 16);
}

// A-fragment (16x32, 16-bit): lane l<16 holds row l, K = 0..7 (elems 0..7) and
// 16..23 (elems 8..15); lane 16+l holds row l, K = 8..15 and 24..31.
__device__ __forceinline__ v16bf load_frag_a(const ushort_t* base, int row_base,
                                             int k0, int ld) {
    int l = threadIdx.x & 31;
    int r = row_base + (l & 15);
    int kk = k0 + ((l >> 4) << 3);         // +8 for upper lane half
    union { v16bf v; uint4 q[2]; } u;
    const ushort_t* p = base + (size_t)r * ld + kk;
    u.q[0] = *reinterpret_cast<const uint4*>(p);       // K .. K+7
    u.q[1] = *reinterpret_cast<const uint4*>(p + 16);  // K+16 .. K+23
    return u.v;
}

// B-fragment (32x16, 16-bit): lane l<16 holds column l, K = 0..7 and 16..23; lane 16+l
// holds column l, K = 8..15 and 24..31.  base is B^T (rows = columns of B), contiguous K.
__device__ __forceinline__ v16bf load_frag_b(const ushort_t* base, int col_base,
                                             int k0, int ld) {
    int l = threadIdx.x & 31;
    int c = col_base + (l & 15);
    int kk = k0 + ((l >> 4) << 3);         // +8 for upper lane half
    union { v16bf v; uint4 q[2]; } u;
    const ushort_t* p = base + (size_t)c * ld + kk;
    u.q[0] = *reinterpret_cast<const uint4*>(p);       // K .. K+7
    u.q[1] = *reinterpret_cast<const uint4*>(p + 16);  // K+16 .. K+23
    return u.v;
}

__device__ __forceinline__ v8f wmma_bf16(v16bf a, v16bf b, v8f c) {
    return __builtin_amdgcn_wmma_f32_16x16x32_bf16(
        false, a, false, b, (short)0, c, false, false);
}

#if HAVE_TDM
// Issue a TDM 2D-tile load: rows x 32 bf16 tile, global row stride = ld elems,
// into LDS at `lds` (contiguous rows of 32 bf16).  D# built per ISA ch.8:
// group0 = {count=1, lds_addr, global_addr[56:0], type=2}
// group1 = {wg_mask=0, data_size=2B, tensor_dim0=32, tensor_dim1=rows,
//           tile_dim0=32, tile_dim1=rows, tile_dim2=0, dim0_stride=ld}
__device__ __forceinline__ void tdm_load_2d(const ushort_t* g, ushort_t* lds,
                                            int rows, int ld) {
    unsigned long long ga = (unsigned long long)(size_t)g;
    unsigned lds_off = (unsigned)(size_t)lds;   // aperture: LDS addr = low 32b
    v4u g0;
    g0[0] = 1u;                                          // count=1
    g0[1] = lds_off;                                     // lds_addr
    g0[2] = (unsigned)(ga & 0xffffffffu);                // global_addr[31:0]
    g0[3] = (unsigned)((ga >> 32) & 0x01ffffffu)         // global_addr[56:32]
          | (2u << 30);                                  // type=2 (image)
    v8i_ g1;
    g1[0] = (int)(1u << 16);                             // data_size=1 (2 bytes)
    g1[1] = (int)(32u << 16);                            // tensor_dim0[15:0]=32
    g1[2] = (int)(((unsigned)rows & 0xffffu) << 16);     // dim0 hi=0 | dim1 lo
    g1[3] = (int)(32u << 16);                            // dim1 hi=0 | tile_dim0=32
    g1[4] = rows & 0xffff;                               // tile_dim1 | tile_dim2=0
    g1[5] = ld;                                          // dim0_stride[31:0]
    g1[6] = 0;                                           // stride hi | dim1_stride lo
    g1[7] = 0;
    v4i_ z4 = {0, 0, 0, 0};
    v8i_ z8 = {0, 0, 0, 0, 0, 0, 0, 0};
    __builtin_amdgcn_tensor_load_to_lds(g0, g1, z4, z4, z8, 0);
}
#else
// Fallback: cooperative global->LDS copy of a rows x 32 bf16 tile (256 thr).
__device__ __forceinline__ void copy_tile_to_lds(const ushort_t* g, ushort_t* lds,
                                                 int rows, int ld) {
    int chunks = rows * 4;                    // 16B chunks per tile
    for (int c = threadIdx.x; c < chunks; c += 256) {
        int r = c >> 2, cc = (c & 3) * 8;
        *reinterpret_cast<uint4*>(lds + r * 32 + cc) =
            *reinterpret_cast<const uint4*>(g + (size_t)r * ld + cc);
    }
}
#endif

// ---------------- weight convert + transpose: W[K,N] f32 -> Wt[N,K] bf16 ----

__global__ __launch_bounds__(256)
void w_to_bf16_t(const float* __restrict__ W, ushort_t* __restrict__ Wt,
                 int K, int N) {
    int idx = blockIdx.x * 256 + threadIdx.x;
    if (idx >= K * N) return;
    int n = idx / K, k = idx - n * K;
    Wt[idx] = f2bf_u(W[(size_t)k * N + n]);
}

// ---------------- layernorm (f32 in, bf16 out), one row per block ----------

__global__ __launch_bounds__(256)
void layernorm_to_bf16(const float* __restrict__ x, const float* __restrict__ g,
                       const float* __restrict__ bta, ushort_t* __restrict__ out) {
    const int D = D_MODEL;
    int row = blockIdx.x, t = threadIdx.x;
    const float* xr = x + (size_t)row * D;
    float s = 0.f, s2 = 0.f;
    for (int j = t; j < D; j += 256) { float v = xr[j]; s += v; s2 += v * v; }
    __shared__ float rs[256], rq[256];
    rs[t] = s; rq[t] = s2; __syncthreads();
    for (int o = 128; o > 0; o >>= 1) {
        if (t < o) { rs[t] += rs[t + o]; rq[t] += rq[t + o]; }
        __syncthreads();
    }
    float mean = rs[0] / D;
    float var  = rq[0] / D - mean * mean;
    float inv  = rsqrtf(var + 1e-5f);
    for (int j = t; j < D; j += 256) {
        float v = (xr[j] - mean) * inv * g[j] + bta[j];
        out[(size_t)row * D + j] = f2bf_u(v);
    }
}

// ---------------- WMMA GEMM with TDM-staged, double-buffered LDS tiles ------
// 256 threads = 8 waves; block tile 64(M) x 64(N); wave tile 16 x 32.
// Per k-step, wave 0 TDM-loads the NEXT 64x32 A and B tiles into the ping-pong
// LDS buffers while all waves compute from the current ones.
// mode: 0 = bf16 row-major, 1 = f32 + residual, 2 = GELU bf16,
//       3/4 = per-head [B,H,S,64] bf16 (Q/K), 5 = per-head [B,H,64,S] bf16 (V^T)

__global__ __launch_bounds__(256)
void gemm_bf16_wmma(const ushort_t* __restrict__ A, const ushort_t* __restrict__ Wt,
                    const float* __restrict__ bias, const float* __restrict__ resid,
                    void* __restrict__ Cout, int M, int N, int K, int mode, int Ssz) {
    __shared__ ushort_t ldsA[2][64 * 32];
    __shared__ ushort_t ldsB[2][64 * 32];

    int w  = threadIdx.x >> 5;
    int wm = w >> 1, wn = w & 1;
    int m0 = blockIdx.y * 64;
    int n0 = blockIdx.x * 64;
    const ushort_t* Ab = A  + (size_t)m0 * K;
    const ushort_t* Bb = Wt + (size_t)n0 * K;

    v8f acc0 = {}, acc1 = {};
    int cur = 0;

#if HAVE_TDM
    if (threadIdx.x < 32) {                       // wave 0 drives the TDM
        tdm_load_2d(Ab, ldsA[0], 64, K);
        tdm_load_2d(Bb, ldsB[0], 64, K);
    }
    for (int k0 = 0; k0 < K; k0 += 32, cur ^= 1) {
        if (threadIdx.x < 32) {
            if (k0 + 32 < K) {
                tdm_load_2d(Ab + k0 + 32, ldsA[cur ^ 1], 64, K);
                tdm_load_2d(Bb + k0 + 32, ldsB[cur ^ 1], 64, K);
                __builtin_amdgcn_s_wait_tensorcnt(2);  // current pair done
            } else {
                __builtin_amdgcn_s_wait_tensorcnt(0);
            }
        }
        __syncthreads();
        v16bf a  = load_frag_a(ldsA[cur], wm * 16,      0, 32);
        v16bf b0 = load_frag_b(ldsB[cur], wn * 32,      0, 32);
        v16bf b1 = load_frag_b(ldsB[cur], wn * 32 + 16, 0, 32);
        acc0 = wmma_bf16(a, b0, acc0);
        acc1 = wmma_bf16(a, b1, acc1);
        __syncthreads();                           // before buffer reuse
    }
#else
    for (int k0 = 0; k0 < K; k0 += 32) {
        copy_tile_to_lds(Ab + k0, ldsA[0], 64, K);
        copy_tile_to_lds(Bb + k0, ldsB[0], 64, K);
        __syncthreads();
        v16bf a  = load_frag_a(ldsA[0], wm * 16,      0, 32);
        v16bf b0 = load_frag_b(ldsB[0], wn * 32,      0, 32);
        v16bf b1 = load_frag_b(ldsB[0], wn * 32 + 16, 0, 32);
        acc0 = wmma_bf16(a, b0, acc0);
        acc1 = wmma_bf16(a, b1, acc1);
        __syncthreads();
    }
#endif

    int l  = threadIdx.x & 31;
    int n  = l & 15;
    int mb = (l >> 4) << 3;
    for (int t = 0; t < 2; ++t) {
        v8f acc = t ? acc1 : acc0;
        int col = n0 + wn * 32 + t * 16 + n;
        float bcol = bias[col];
#pragma unroll
        for (int i = 0; i < 8; ++i) {
            int row = m0 + wm * 16 + mb + i;
            float v = acc[i] + bcol;
            if (mode == 1) {
                ((float*)Cout)[(size_t)row * N + col] =
                    v + resid[(size_t)row * N + col];
            } else if (mode == 2) {
                float ge = 0.5f * v * (1.0f + erff(v * 0.70710678118f));
                ((ushort_t*)Cout)[(size_t)row * N + col] = f2bf_u(ge);
            } else if (mode == 0) {
                ((ushort_t*)Cout)[(size_t)row * N + col] = f2bf_u(v);
            } else {
                int b = row / Ssz, s = row - b * Ssz;
                int h = col >> 6,  d = col & 63;
                size_t oidx = (mode == 5)
                    ? ((size_t)(b * N_HEADS + h) * D_K + d) * Ssz + s
                    : ((size_t)(b * N_HEADS + h) * Ssz + s) * D_K + d;
                ((ushort_t*)Cout)[oidx] = f2bf_u(v);
            }
        }
    }
}

// ---------------- flash attention with fused geo-bias gather ----------------
// 1 wave per (b, h, 16-query tile); 32 keys per iteration; scores never spilled.

__global__ __launch_bounds__(32)
void attn_flash_wmma(const ushort_t* __restrict__ Q,   // [B,H,S,64] bf16
                     const ushort_t* __restrict__ Kt,  // [B,H,S,64] bf16
                     const ushort_t* __restrict__ Vt,  // [B,H,64,S] bf16
                     const int* __restrict__ geo,      // [B,3,S,S]
                     const unsigned char* __restrict__ mask, // [B,S]
                     const float* __restrict__ v_off,  // [8,12]
                     const float* __restrict__ h_off,
                     const float* __restrict__ size_b,
                     ushort_t* __restrict__ Out,       // [B,S,768] bf16
                     int S) {
    int q0 = blockIdx.x * 16;
    int h  = blockIdx.y;
    int b  = blockIdx.z;
    int l  = threadIdx.x;

    __shared__ float tv[8], th_[8], ts_[8];
    __shared__ ushort_t p_lds[16 * 32];
    if (l < 8) {
        tv[l]  = v_off[l * N_HEADS + h];
        th_[l] = h_off[l * N_HEADS + h];
        ts_[l] = size_b[l * N_HEADS + h];
    }
    __syncthreads();

    size_t bh = (size_t)(b * N_HEADS + h);
    const ushort_t* Qb    = Q  + bh * S * D_K;
    const ushort_t* Kbase = Kt + bh * S * D_K;
    const ushort_t* Vbase = Vt + bh * D_K * S;
    const int* gA = geo + (size_t)(b * 3 + 0) * S * S;
    const int* gB = geo + (size_t)(b * 3 + 1) * S * S;
    const int* gC = geo + (size_t)(b * 3 + 2) * S * S;
    const unsigned char* mk = mask + (size_t)b * S;

    v16bf qa0 = load_frag_a(Qb, q0, 0,  D_K);
    v16bf qa1 = load_frag_a(Qb, q0, 32, D_K);

    v8f o0 = {}, o1 = {}, o2 = {}, o3 = {};
    float mrow[8], lrow[8];
#pragma unroll
    for (int i = 0; i < 8; ++i) { mrow[i] = -3.0e38f; lrow[i] = 0.f; }

    int n  = l & 15;
    int mb = (l >> 4) << 3;

    for (int kb = 0; kb < S; kb += 32) {
        // ---- S = Q K^T / sqrt(d) ----
        v16bf k00 = load_frag_b(Kbase, kb,      0,  D_K);
        v16bf k01 = load_frag_b(Kbase, kb,      32, D_K);
        v16bf k10 = load_frag_b(Kbase, kb + 16, 0,  D_K);
        v16bf k11 = load_frag_b(Kbase, kb + 16, 32, D_K);
        v8f s0 = {}; s0 = wmma_bf16(qa0, k00, s0); s0 = wmma_bf16(qa1, k01, s0);
        v8f s1 = {}; s1 = wmma_bf16(qa0, k10, s1); s1 = wmma_bf16(qa1, k11, s1);

        // ---- fused geo-bias gather + mask + online softmax ----
        int kk0 = kb + n, kk1 = kb + 16 + n;
        bool m0b = mk[kk0] != 0, m1b = mk[kk1] != 0;
#pragma unroll
        for (int i = 0; i < 8; ++i) {
            int q = q0 + mb + i;
            size_t ro = (size_t)q * S;
            float bias0 = tv[gA[ro + kk0]] + th_[gB[ro + kk0]] + ts_[gC[ro + kk0]];
            float bias1 = tv[gA[ro + kk1]] + th_[gB[ro + kk1]] + ts_[gC[ro + kk1]];
            float x0 = s0[i] * 0.125f + bias0; if (m0b) x0 = -3.0e38f;
            float x1 = s1[i] * 0.125f + bias1; if (m1b) x1 = -3.0e38f;
            float mx = fmaxf(x0, x1);
            for (int d = 1; d < 16; d <<= 1) mx = fmaxf(mx, __shfl_xor(mx, d, 32));
            float mnew  = fmaxf(mrow[i], mx);
            float alpha = __expf(mrow[i] - mnew);
            mrow[i] = mnew;
            x0 = __expf(x0 - mnew);
            x1 = __expf(x1 - mnew);
            float rs = x0 + x1;
            for (int d = 1; d < 16; d <<= 1) rs += __shfl_xor(rs, d, 32);
            lrow[i] = lrow[i] * alpha + rs;
            o0[i] *= alpha; o1[i] *= alpha; o2[i] *= alpha; o3[i] *= alpha;
            s0[i] = x0; s1[i] = x1;
        }

        // ---- re-fragment P via LDS, then O += P V ----
#pragma unroll
        for (int i = 0; i < 8; ++i) {
            p_lds[(mb + i) * 32 + n]      = f2bf_u(s0[i]);
            p_lds[(mb + i) * 32 + 16 + n] = f2bf_u(s1[i]);
        }
        __syncthreads();
        v16bf pf  = load_frag_a(p_lds, 0, 0, 32);
        v16bf vf0 = load_frag_b(Vbase, 0,  kb, S);
        v16bf vf1 = load_frag_b(Vbase, 16, kb, S);
        v16bf vf2 = load_frag_b(Vbase, 32, kb, S);
        v16bf vf3 = load_frag_b(Vbase, 48, kb, S);
        o0 = wmma_bf16(pf, vf0, o0);
        o1 = wmma_bf16(pf, vf1, o1);
        o2 = wmma_bf16(pf, vf2, o2);
        o3 = wmma_bf16(pf, vf3, o3);
        __syncthreads();
    }

#pragma unroll
    for (int i = 0; i < 8; ++i) {
        float inv = 1.0f / lrow[i];
        int q = q0 + mb + i;
        size_t base = ((size_t)b * S + q) * D_MODEL + h * D_K;
        Out[base + 0  + n] = f2bf_u(o0[i] * inv);
        Out[base + 16 + n] = f2bf_u(o1[i] * inv);
        Out[base + 32 + n] = f2bf_u(o2[i] * inv);
        Out[base + 48 + n] = f2bf_u(o3[i] * inv);
    }
}

// ---------------- orchestration ----------------

extern "C" void kernel_launch(void* const* d_in, const int* in_sizes, int n_in,
                              void* d_out, int out_size, void* d_ws, size_t ws_size,
                              hipStream_t stream) {
    const int B = 2, S = 2048, M = B * S;

    const float* x      = (const float*)d_in[0];
    const int*   geo    = (const int*)d_in[1];
    const unsigned char* mask = (const unsigned char*)d_in[2];
    const float* wq = (const float*)d_in[3],  * bq = (const float*)d_in[4];
    const float* wk = (const float*)d_in[5],  * bk = (const float*)d_in[6];
    const float* wv = (const float*)d_in[7],  * bv = (const float*)d_in[8];
    const float* wo = (const float*)d_in[9],  * bo = (const float*)d_in[10];
    const float* voff = (const float*)d_in[11];
    const float* hoff = (const float*)d_in[12];
    const float* szb  = (const float*)d_in[13];
    const float* g1 = (const float*)d_in[14], * b1 = (const float*)d_in[15];
    const float* g2 = (const float*)d_in[16], * b2 = (const float*)d_in[17];
    const float* wff1 = (const float*)d_in[18], * bff1 = (const float*)d_in[19];
    const float* wff2 = (const float*)d_in[20], * bff2 = (const float*)d_in[21];

    // workspace carve
    char* wsp = (char*)d_ws;
    size_t off = 0;
    auto carve = [&](size_t bytes) -> void* {
        void* p = wsp + off;
        off = (off + bytes + 255) & ~(size_t)255;
        return p;
    };
    ushort_t* wq_t  = (ushort_t*)carve((size_t)D_MODEL * D_MODEL * 2);
    ushort_t* wk_t  = (ushort_t*)carve((size_t)D_MODEL * D_MODEL * 2);
    ushort_t* wv_t  = (ushort_t*)carve((size_t)D_MODEL * D_MODEL * 2);
    ushort_t* wo_t  = (ushort_t*)carve((size_t)D_MODEL * D_MODEL * 2);
    ushort_t* ff1_t = (ushort_t*)carve((size_t)D_MODEL * D_FF * 2);
    ushort_t* ff2_t = (ushort_t*)carve((size_t)D_MODEL * D_FF * 2);
    ushort_t* xn1   = (ushort_t*)carve((size_t)M * D_MODEL * 2);
    ushort_t* Qb    = (ushort_t*)carve((size_t)M * D_MODEL * 2);
    ushort_t* Kb    = (ushort_t*)carve((size_t)M * D_MODEL * 2);
    ushort_t* Vtb   = (ushort_t*)carve((size_t)M * D_MODEL * 2);
    ushort_t* attn  = (ushort_t*)carve((size_t)M * D_MODEL * 2);
    float*    x2    = (float*)   carve((size_t)M * D_MODEL * 4);
    ushort_t* xn2   = (ushort_t*)carve((size_t)M * D_MODEL * 2);
    ushort_t* hbf   = (ushort_t*)carve((size_t)M * D_FF * 2);

    // 1) weight convert + transpose
    int eW  = D_MODEL * D_MODEL, eF = D_MODEL * D_FF;
    w_to_bf16_t<<<(eW + 255) / 256, 256, 0, stream>>>(wq,   wq_t,  D_MODEL, D_MODEL);
    w_to_bf16_t<<<(eW + 255) / 256, 256, 0, stream>>>(wk,   wk_t,  D_MODEL, D_MODEL);
    w_to_bf16_t<<<(eW + 255) / 256, 256, 0, stream>>>(wv,   wv_t,  D_MODEL, D_MODEL);
    w_to_bf16_t<<<(eW + 255) / 256, 256, 0, stream>>>(wo,   wo_t,  D_MODEL, D_MODEL);
    w_to_bf16_t<<<(eF + 255) / 256, 256, 0, stream>>>(wff1, ff1_t, D_MODEL, D_FF);
    w_to_bf16_t<<<(eF + 255) / 256, 256, 0, stream>>>(wff2, ff2_t, D_FF,    D_MODEL);

    // 2) LN1
    layernorm_to_bf16<<<M, 256, 0, stream>>>(x, g1, b1, xn1);

    // 3) QKV projections (fused head-layout epilogues)
    dim3 gQ(D_MODEL / 64, M / 64);
    gemm_bf16_wmma<<<gQ, 256, 0, stream>>>(xn1, wq_t, bq, nullptr, Qb,
                                           M, D_MODEL, D_MODEL, 3, S);
    gemm_bf16_wmma<<<gQ, 256, 0, stream>>>(xn1, wk_t, bk, nullptr, Kb,
                                           M, D_MODEL, D_MODEL, 4, S);
    gemm_bf16_wmma<<<gQ, 256, 0, stream>>>(xn1, wv_t, bv, nullptr, Vtb,
                                           M, D_MODEL, D_MODEL, 5, S);

    // 4) flash attention with fused geo-bias
    dim3 gA(S / 16, N_HEADS, B);
    attn_flash_wmma<<<gA, 32, 0, stream>>>(Qb, Kb, Vtb, geo, mask,
                                           voff, hoff, szb, attn, S);

    // 5) output projection + residual (f32)
    gemm_bf16_wmma<<<gQ, 256, 0, stream>>>(attn, wo_t, bo, x, x2,
                                           M, D_MODEL, D_MODEL, 1, S);

    // 6) LN2
    layernorm_to_bf16<<<M, 256, 0, stream>>>(x2, g2, b2, xn2);

    // 7) FF1 + exact GELU (bf16)
    dim3 gF(D_FF / 64, M / 64);
    gemm_bf16_wmma<<<gF, 256, 0, stream>>>(xn2, ff1_t, bff1, nullptr, hbf,
                                           M, D_FF, D_MODEL, 2, S);

    // 8) FF2 + residual -> d_out (f32)
    gemm_bf16_wmma<<<gQ, 256, 0, stream>>>(hbf, ff2_t, bff2, x2, (float*)d_out,
                                           M, D_MODEL, D_FF, 1, S);
}
